// NonOverlappingFlatVQVAE_51874615001585
// MI455X (gfx1250) — matrix operand order repaired and checked by reading, not compiled, not run on hardware
//
#include <hip/hip_runtime.h>

typedef __bf16 bf16;
typedef __attribute__((ext_vector_type(16))) __bf16 v16bf;
typedef __attribute__((ext_vector_type(8)))  float  v8f;

// ---------------------------------------------------------------------------
// Problem constants (from reference setup_inputs)
// B=32, Cin=3, H=W=256, P=16 -> Npatch = 8192; C=256, D=512, Kcb=8192, F=768
// ---------------------------------------------------------------------------
static constexpr int MROWS = 8192;
static constexpr int FEAT  = 768;
static constexpr int CCH   = 256;
static constexpr int DDIM  = 512;
static constexpr int KCB   = 8192;

// ---------------------------------------------------------------------------
// CDNA5 async global->LDS copy (GLOBAL_LOAD_ASYNC_TO_LDS_B128, ASYNCcnt)
// VDST = VGPR holding LDS byte address, VADDR = 64-bit global address.
// ---------------------------------------------------------------------------
__device__ __forceinline__ unsigned lds_addr(const void* p) {
  return (unsigned)(unsigned long long)(__attribute__((address_space(3))) const char*)p;
}
#define ASYNC_COPY_B128(loff, gptr)                                        \
  asm volatile("global_load_async_to_lds_b128 %0, %1, off"                 \
               :: "v"(loff), "v"(gptr) : "memory")
#define ASYNC_WAIT() asm volatile("s_wait_asynccnt 0x0" ::: "memory")

// ---------------------------------------------------------------------------
// Small prep kernels
// ---------------------------------------------------------------------------
__global__ __launch_bounds__(256) void patchify_kernel(const float* __restrict__ x,
                                                       bf16* __restrict__ A) {
  int t = blockIdx.x * 256 + threadIdx.x;           // < 8192*768
  int row = t / FEAT, col = t - row * FEAT;
  int b = row >> 8, ij = row & 255, i = ij >> 4, j = ij & 15;
  int c = col >> 8, pq = col & 255, p = pq >> 4, q = pq & 15;
  int xIdx = ((b * 3 + c) << 16) + ((i * 16 + p) << 8) + (j * 16 + q);
  A[t] = (bf16)x[xIdx];
}

__global__ __launch_bounds__(256) void tconv_kernel(const float* __restrict__ in,
                                                    bf16* __restrict__ out,
                                                    int R, int S) {
  int t = blockIdx.x * 256 + threadIdx.x;
  if (t >= R * S) return;
  int r = t / S, s = t - r * S;
  out[(size_t)s * R + r] = (bf16)in[t];
}

__global__ __launch_bounds__(256) void conv_kernel(const float* __restrict__ in,
                                                   bf16* __restrict__ out, int n) {
  int t = blockIdx.x * 256 + threadIdx.x;
  if (t < n) out[t] = (bf16)in[t];
}

__global__ __launch_bounds__(256) void expand_upb_kernel(const float* __restrict__ upb,
                                                         float* __restrict__ outb) {
  int t = blockIdx.x * 256 + threadIdx.x;
  if (t < FEAT) outb[t] = upb[t >> 8];
}

__global__ void zero_kernel(float* p) { *p = 0.f; }

__global__ void finalize_diff_kernel(const float* acc, float* out) {
  *out = *acc * (1.0f / ((float)MROWS * (float)DDIM));
}

// ---------------------------------------------------------------------------
// bf16 WMMA GEMM: C[M,N] = act(A[M,K] * B[K,N] + bias[N])
// Block tile 128x64, 8 waves (4 M x 2 N), wave tile 32x32 (2x2 WMMA 16x16x32)
// A tile staged with async global->LDS; B tile staged with VGPR transpose.
// ---------------------------------------------------------------------------
__global__ __launch_bounds__(256) void gemm_bf16_kernel(
    const bf16* __restrict__ A, const bf16* __restrict__ B,
    const float* __restrict__ bias, bf16* __restrict__ outBf,
    float* __restrict__ outF, int M, int N, int K, int relu) {
  __shared__ bf16 As[128][32];   // 8 KB
  __shared__ bf16 Bs[64][32];    // 4 KB, N-major (transposed tile)

  const int tid  = threadIdx.x;
  const int wave = tid >> 5, lane = tid & 31;
  const int wm = wave >> 1, wn = wave & 1;
  const int half = lane >> 4, l16 = lane & 15;
  const int m0 = blockIdx.y * 128, n0 = blockIdx.x * 64;

  const v8f vzero = {0.f, 0.f, 0.f, 0.f, 0.f, 0.f, 0.f, 0.f};
  v8f acc[2][2];
  acc[0][0] = vzero; acc[0][1] = vzero; acc[1][0] = vzero; acc[1][1] = vzero;

  for (int k0 = 0; k0 < K; k0 += 32) {
    __syncthreads();
    if (k0 + 32 < K) {
      __builtin_prefetch(A + (size_t)(m0 + (tid & 127)) * K + k0 + 32, 0, 1);
      __builtin_prefetch(B + (size_t)(k0 + 32 + (tid & 31)) * N + n0, 0, 1);
    }
    // A tile: 128x32 bf16 = 512 x b128, 2 async copies per thread
#pragma unroll
    for (int i = 0; i < 2; ++i) {
      int idx = tid + i * 256;
      int r = idx >> 2, c = idx & 3;
      ASYNC_COPY_B128(lds_addr(&As[r][c * 8]),
                      A + (size_t)(m0 + r) * K + k0 + c * 8);
    }
    // B tile: 32x64 bf16 loaded row-wise, stored N-major (transpose in VGPRs)
    {
      int r = tid >> 3;              // 0..31
      int cs = (tid & 7) << 3;       // 0..56
      uint4 v = *(const uint4*)(B + (size_t)(k0 + r) * N + n0 + cs);
      const bf16* pv = (const bf16*)&v;
#pragma unroll
      for (int j = 0; j < 8; ++j) Bs[cs + j][r] = pv[j];
    }
    ASYNC_WAIT();
    __syncthreads();

    // fragments per ISA 7.12.2 16-bit layouts
    v16bf af[2], bfr[2];
#pragma unroll
    for (int t2 = 0; t2 < 2; ++t2) {
      int mr = wm * 32 + t2 * 16 + l16;
      v16bf a;
      ((uint4*)&a)[0] = *(const uint4*)&As[mr][half * 8];
      ((uint4*)&a)[1] = *(const uint4*)&As[mr][16 + half * 8];
      af[t2] = a;
      int nr = wn * 32 + t2 * 16 + l16;
      v16bf b;
      ((uint4*)&b)[0] = *(const uint4*)&Bs[nr][half * 8];
      ((uint4*)&b)[1] = *(const uint4*)&Bs[nr][16 + half * 8];
      bfr[t2] = b;
    }
#pragma unroll
    for (int tm = 0; tm < 2; ++tm)
#pragma unroll
      for (int tn = 0; tn < 2; ++tn)
        acc[tm][tn] = __builtin_amdgcn_wmma_f32_16x16x32_bf16(
            false, af[tm], false, bfr[tn], (short)0, acc[tm][tn], false, false);
  }

  // epilogue: C layout M = r + 8*half, N = l16
#pragma unroll
  for (int tm = 0; tm < 2; ++tm)
#pragma unroll
    for (int tn = 0; tn < 2; ++tn) {
      int n = n0 + wn * 32 + tn * 16 + l16;
      float bb = bias ? bias[n] : 0.f;
#pragma unroll
      for (int r = 0; r < 8; ++r) {
        int m = m0 + wm * 32 + tm * 16 + half * 8 + r;
        float v = acc[tm][tn][r] + bb;
        if (relu) v = v > 0.f ? v : 0.f;
        if (outF)  outF[(size_t)m * N + n] = v;
        if (outBf) outBf[(size_t)m * N + n] = (bf16)v;
      }
    }
}

// ---------------------------------------------------------------------------
// codebook row norms
// ---------------------------------------------------------------------------
__global__ __launch_bounds__(256) void cnorm_kernel(const float* __restrict__ cb,
                                                    float* __restrict__ cnorm) {
  int k = blockIdx.x, tid = threadIdx.x;
  float s = 0.f;
  for (int d = tid; d < DDIM; d += 256) {
    float v = cb[(size_t)k * DDIM + d];
    s += v * v;
  }
#pragma unroll
  for (int m = 16; m >= 1; m >>= 1) s += __shfl_xor(s, m, 32);
  __shared__ float red[8];
  if ((tid & 31) == 0) red[tid >> 5] = s;
  __syncthreads();
  if (tid < 8) {
    float v = red[tid];
#pragma unroll
    for (int m = 4; m >= 1; m >>= 1) v += __shfl_xor(v, m, 32);
    if (tid == 0) cnorm[k] = v;
  }
}

// ---------------------------------------------------------------------------
// Fused VQ: scores = z . cbT via WMMA, dist = |c|^2 - 2*score, running argmin.
// Never materializes the 8192x8192 (256 MB) score matrix.
// The 128x512 z block (128 KB) is LDS-resident for the whole kernel (CDNA5
// WGP has 320 KB LDS) so z is read from HBM exactly once; only the codebook
// streams through the inner loop.
// ---------------------------------------------------------------------------
__global__ __launch_bounds__(256) void vq_argmin_kernel(
    const bf16* __restrict__ zb, const bf16* __restrict__ cbT,
    const float* __restrict__ cnorm, int* __restrict__ idxOut) {
  __shared__ bf16 zs[128][512];   // 128 KB, resident z block
  __shared__ bf16 Bs[64][32];     // 4 KB code-tile k-slice, N-major
  __shared__ float sbv[128][2];
  __shared__ int sbi[128][2];

  const int tid  = threadIdx.x;
  const int wave = tid >> 5, lane = tid & 31;
  const int wm = wave >> 1, wn = wave & 1;
  const int half = lane >> 4, l16 = lane & 15;
  const int m0 = blockIdx.x * 128;

  const v8f vzero = {0.f, 0.f, 0.f, 0.f, 0.f, 0.f, 0.f, 0.f};

  // async-copy the whole z block: 128*512 bf16 = 8192 x b128, 32 per thread
  {
    const bf16* src = zb + (size_t)m0 * DDIM;
    bf16* dst = &zs[0][0];
#pragma unroll 4
    for (int i = tid; i < 8192; i += 256) {
      ASYNC_COPY_B128(lds_addr(dst + (size_t)i * 8), src + (size_t)i * 8);
    }
  }
  ASYNC_WAIT();
  __syncthreads();

  float best[2][2][8];
  int   ibst[2][2][8];
#pragma unroll
  for (int a = 0; a < 2; ++a)
#pragma unroll
    for (int b = 0; b < 2; ++b)
#pragma unroll
      for (int r = 0; r < 8; ++r) { best[a][b][r] = 3.4e38f; ibst[a][b][r] = 0; }

  const int br = tid >> 3;          // B-tile row this thread loads
  const int bcs = (tid & 7) << 3;   // B-tile col segment

  for (int c0 = 0; c0 < KCB; c0 += 64) {
    v8f acc[2][2];
    acc[0][0] = vzero; acc[0][1] = vzero; acc[1][0] = vzero; acc[1][1] = vzero;

#pragma unroll 2
    for (int ks = 0; ks < DDIM; ks += 32) {
      __syncthreads();
      {
        uint4 v = *(const uint4*)(cbT + (size_t)(ks + br) * KCB + c0 + bcs);
        const bf16* pv = (const bf16*)&v;
#pragma unroll
        for (int j = 0; j < 8; ++j) Bs[bcs + j][br] = pv[j];
      }
      __syncthreads();

      v16bf af[2], bfr[2];
#pragma unroll
      for (int t2 = 0; t2 < 2; ++t2) {
        int mr = wm * 32 + t2 * 16 + l16;
        v16bf a;
        ((uint4*)&a)[0] = *(const uint4*)&zs[mr][ks + half * 8];
        ((uint4*)&a)[1] = *(const uint4*)&zs[mr][ks + 16 + half * 8];
        af[t2] = a;
        int nr = wn * 32 + t2 * 16 + l16;
        v16bf b;
        ((uint4*)&b)[0] = *(const uint4*)&Bs[nr][half * 8];
        ((uint4*)&b)[1] = *(const uint4*)&Bs[nr][16 + half * 8];
        bfr[t2] = b;
      }
#pragma unroll
      for (int tm = 0; tm < 2; ++tm)
#pragma unroll
        for (int tn = 0; tn < 2; ++tn)
          acc[tm][tn] = __builtin_amdgcn_wmma_f32_16x16x32_bf16(
              false, af[tm], false, bfr[tn], (short)0, acc[tm][tn], false, false);
    }

    // dist = |c|^2 - 2*dot ; update running argmin (|z|^2 constant per row)
#pragma unroll
    for (int tm = 0; tm < 2; ++tm)
#pragma unroll
      for (int tn = 0; tn < 2; ++tn) {
        int code = c0 + wn * 32 + tn * 16 + l16;
        float cn = cnorm[code];
#pragma unroll
        for (int r = 0; r < 8; ++r) {
          float d = cn - 2.0f * acc[tm][tn][r];
          if (d < best[tm][tn][r]) { best[tm][tn][r] = d; ibst[tm][tn][r] = code; }
        }
      }
  }

  // reduce argmin: across tn, then across the 16 lanes of each half, then waves
#pragma unroll
  for (int tm = 0; tm < 2; ++tm)
#pragma unroll
    for (int r = 0; r < 8; ++r) {
      float bv = best[tm][0][r];
      int bi = ibst[tm][0][r];
      if (best[tm][1][r] < bv || (best[tm][1][r] == bv && ibst[tm][1][r] < bi)) {
        bv = best[tm][1][r]; bi = ibst[tm][1][r];
      }
#pragma unroll
      for (int mask = 1; mask <= 8; mask <<= 1) {
        float ov = __shfl_xor(bv, mask, 32);
        int   oi = __shfl_xor(bi, mask, 32);
        if (ov < bv || (ov == bv && oi < bi)) { bv = ov; bi = oi; }
      }
      if (l16 == 0) {
        int rl = wm * 32 + tm * 16 + half * 8 + r;
        sbv[rl][wn] = bv;
        sbi[rl][wn] = bi;
      }
    }
  __syncthreads();
  if (tid < 128) {
    float v0 = sbv[tid][0]; int i0 = sbi[tid][0];
    float v1 = sbv[tid][1]; int i1 = sbi[tid][1];
    idxOut[m0 + tid] = (v1 < v0 || (v1 == v0 && i1 < i0)) ? i1 : i0;
  }
}

// ---------------------------------------------------------------------------
// gather q = codebook[idx], commitment-loss partial sums, q -> bf16
// ---------------------------------------------------------------------------
__global__ __launch_bounds__(256) void gather_q_diff_kernel(
    const float* __restrict__ cb, const int* __restrict__ idx,
    const float* __restrict__ zf, bf16* __restrict__ qbf,
    float* __restrict__ acc) {
  int t = blockIdx.x * 256 + threadIdx.x;  // < 8192*512
  int n = t >> 9, d = t & 511;
  float q = cb[(size_t)idx[n] * DDIM + d];
  float df = q - zf[t];
  qbf[t] = (bf16)q;
  float s = df * df;
#pragma unroll
  for (int m = 16; m >= 1; m >>= 1) s += __shfl_xor(s, m, 32);
  __shared__ float red[8];
  if ((threadIdx.x & 31) == 0) red[threadIdx.x >> 5] = s;
  __syncthreads();
  if (threadIdx.x < 8) {
    float v = red[threadIdx.x];
#pragma unroll
    for (int m = 4; m >= 1; m >>= 1) v += __shfl_xor(v, m, 32);
    if (threadIdx.x == 0) atomicAdd(acc, v);
  }
}

// ---------------------------------------------------------------------------
// scatter [patch, (o,p,q)] -> NCHW output
// ---------------------------------------------------------------------------
__global__ __launch_bounds__(256) void scatter_dec_kernel(const float* __restrict__ decf,
                                                          float* __restrict__ out) {
  int t = blockIdx.x * 256 + threadIdx.x;  // < 8192*768
  int row = t / FEAT, col = t - row * FEAT;
  int b = row >> 8, ij = row & 255, i = ij >> 4, j = ij & 15;
  int o = col >> 8, pq = col & 255, p = pq >> 4, q = pq & 15;
  int outIdx = ((b * 3 + o) << 16) + ((i * 16 + p) << 8) + (j * 16 + q);
  out[outIdx] = decf[t];
}

// ---------------------------------------------------------------------------
// launch
// ---------------------------------------------------------------------------
extern "C" void kernel_launch(void* const* d_in, const int* in_sizes, int n_in,
                              void* d_out, int out_size, void* d_ws, size_t ws_size,
                              hipStream_t stream) {
  const float* x        = (const float*)d_in[0];
  const float* pe_w     = (const float*)d_in[1];
  const float* pe_b     = (const float*)d_in[2];
  const float* mix1_w   = (const float*)d_in[3];
  const float* mix1_b   = (const float*)d_in[4];
  const float* mix2_w   = (const float*)d_in[5];
  const float* mix2_b   = (const float*)d_in[6];
  const float* qc_w     = (const float*)d_in[7];
  const float* qc_b     = (const float*)d_in[8];
  const float* codebook = (const float*)d_in[9];
  const float* dpre1_w  = (const float*)d_in[10];
  const float* dpre1_b  = (const float*)d_in[11];
  const float* dpre2_w  = (const float*)d_in[12];
  const float* dpre2_b  = (const float*)d_in[13];
  const float* up_w     = (const float*)d_in[14];
  const float* up_b     = (const float*)d_in[15];
  (void)in_sizes; (void)n_in; (void)out_size; (void)ws_size;

  float* out = (float*)d_out;
  char* ws = (char*)d_ws;
  constexpr size_t DEC = (size_t)MROWS * FEAT;  // 6291456 floats

  size_t off = 0;
  auto alloc = [&](size_t bytes) -> void* {
    void* p = ws + off;
    off += (bytes + 255) & ~(size_t)255;
    return p;
  };
  bf16* A_pe    = (bf16*)alloc((size_t)MROWS * FEAT * 2);
  bf16* B_pe    = (bf16*)alloc((size_t)FEAT * CCH * 2);
  bf16* B_mix1  = (bf16*)alloc((size_t)CCH * CCH * 2);
  bf16* B_mix2  = (bf16*)alloc((size_t)CCH * CCH * 2);
  bf16* B_qc    = (bf16*)alloc((size_t)CCH * DDIM * 2);
  bf16* B_dpre1 = (bf16*)alloc((size_t)DDIM * CCH * 2);
  bf16* B_dpre2 = (bf16*)alloc((size_t)CCH * CCH * 2);
  bf16* B_up    = (bf16*)alloc((size_t)CCH * FEAT * 2);
  bf16* cbT     = (bf16*)alloc((size_t)DDIM * KCB * 2);
  float* upb_x  = (float*)alloc((size_t)FEAT * 4);
  bf16* h1      = (bf16*)alloc((size_t)MROWS * CCH * 2);
  bf16* h2      = (bf16*)alloc((size_t)MROWS * CCH * 2);
  bf16* h3      = (bf16*)alloc((size_t)MROWS * CCH * 2);
  float* zf     = (float*)alloc((size_t)MROWS * DDIM * 4);
  bf16* zbf     = (bf16*)alloc((size_t)MROWS * DDIM * 2);
  float* cnorm  = (float*)alloc((size_t)KCB * 4);
  bf16* qbf     = (bf16*)alloc((size_t)MROWS * DDIM * 2);
  bf16* d1      = (bf16*)alloc((size_t)MROWS * CCH * 2);
  bf16* d2      = (bf16*)alloc((size_t)MROWS * CCH * 2);
  float* decf   = (float*)alloc((size_t)MROWS * FEAT * 4);
  float* diffa  = (float*)alloc(256);

  int* idxOut = (int*)(out + DEC + 1);

  // --- prep: patchify + weight convert/transpose to bf16 [K,N] ---
  patchify_kernel<<<MROWS * FEAT / 256, 256, 0, stream>>>(x, A_pe);
  tconv_kernel<<<(CCH * FEAT + 255) / 256, 256, 0, stream>>>(pe_w, B_pe, CCH, FEAT);
  tconv_kernel<<<(CCH * CCH + 255) / 256, 256, 0, stream>>>(mix1_w, B_mix1, CCH, CCH);
  tconv_kernel<<<(CCH * CCH + 255) / 256, 256, 0, stream>>>(mix2_w, B_mix2, CCH, CCH);
  tconv_kernel<<<(DDIM * CCH + 255) / 256, 256, 0, stream>>>(qc_w, B_qc, DDIM, CCH);
  tconv_kernel<<<(CCH * DDIM + 255) / 256, 256, 0, stream>>>(dpre1_w, B_dpre1, CCH, DDIM);
  tconv_kernel<<<(CCH * CCH + 255) / 256, 256, 0, stream>>>(dpre2_w, B_dpre2, CCH, CCH);
  tconv_kernel<<<(KCB * DDIM + 255) / 256, 256, 0, stream>>>(codebook, cbT, KCB, DDIM);
  conv_kernel<<<(CCH * FEAT + 255) / 256, 256, 0, stream>>>(up_w, B_up, CCH * FEAT);
  expand_upb_kernel<<<3, 256, 0, stream>>>(up_b, upb_x);

  // --- encoder GEMM chain (bf16 WMMA, f32 accumulate) ---
  gemm_bf16_kernel<<<dim3(CCH / 64, MROWS / 128), 256, 0, stream>>>(
      A_pe, B_pe, pe_b, h1, nullptr, MROWS, CCH, FEAT, 1);
  gemm_bf16_kernel<<<dim3(CCH / 64, MROWS / 128), 256, 0, stream>>>(
      h1, B_mix1, mix1_b, h2, nullptr, MROWS, CCH, CCH, 1);
  gemm_bf16_kernel<<<dim3(CCH / 64, MROWS / 128), 256, 0, stream>>>(
      h2, B_mix2, mix2_b, h3, nullptr, MROWS, CCH, CCH, 0);
  gemm_bf16_kernel<<<dim3(DDIM / 64, MROWS / 128), 256, 0, stream>>>(
      h3, B_qc, qc_b, zbf, zf, MROWS, DDIM, CCH, 0);

  // --- VQ: fused distance GEMM + argmin (no 256MB score matrix) ---
  cnorm_kernel<<<KCB, 256, 0, stream>>>(codebook, cnorm);
  vq_argmin_kernel<<<MROWS / 128, 256, 0, stream>>>(zbf, cbT, cnorm, idxOut);

  // --- commitment loss + gather quantized vectors ---
  zero_kernel<<<1, 1, 0, stream>>>(diffa);
  gather_q_diff_kernel<<<MROWS * DDIM / 256, 256, 0, stream>>>(codebook, idxOut, zf, qbf, diffa);
  finalize_diff_kernel<<<1, 1, 0, stream>>>(diffa, out + DEC);

  // --- decoder GEMM chain + unpatchify ---
  gemm_bf16_kernel<<<dim3(CCH / 64, MROWS / 128), 256, 0, stream>>>(
      qbf, B_dpre1, dpre1_b, d1, nullptr, MROWS, CCH, DDIM, 1);
  gemm_bf16_kernel<<<dim3(CCH / 64, MROWS / 128), 256, 0, stream>>>(
      d1, B_dpre2, dpre2_b, d2, nullptr, MROWS, CCH, CCH, 1);
  gemm_bf16_kernel<<<dim3(FEAT / 64, MROWS / 128), 256, 0, stream>>>(
      d2, B_up, upb_x, nullptr, decf, MROWS, FEAT, CCH, 0);
  scatter_dec_kernel<<<MROWS * FEAT / 256, 256, 0, stream>>>(decf, out);
}
